// NnqlmCnnBasedLstm_73770358276978
// MI455X (gfx1250) — matrix operand order repaired and checked by reading, not compiled, not run on hardware
//
#include <hip/hip_runtime.h>
#include <hip/hip_bf16.h>

// ---------------------------------------------------------------------------
// Problem constants (match reference)
// ---------------------------------------------------------------------------
constexpr int Bsz  = 32;
constexpr int Lseq = 40;
constexpr int Dd   = 128;          // feature dim; density is Dd x Dd
constexpr int NLay = 2;

typedef __attribute__((ext_vector_type(16))) _Float16 v16h;
typedef __attribute__((ext_vector_type(8)))  float    v8f;

// Branch-free transcendentals on native v_exp_f32 / v_rcp_f32.
__device__ __forceinline__ float sigmoid_f(float x) {
    return __builtin_amdgcn_rcpf(1.0f + __builtin_amdgcn_exp2f(-1.44269504f * x));
}
__device__ __forceinline__ float tanh_f(float x) {
    const float a = __builtin_amdgcn_exp2f(-2.88539008f * __builtin_fabsf(x));
    const float r = (1.0f - a) * __builtin_amdgcn_rcpf(1.0f + a);
    return __builtin_copysignf(r, x);
}

// ---------------------------------------------------------------------------
// Kernel 1: density = outer(xe, xe) / (dot(xe,xe) + 1e-4), stored f16.
// One block per (seq, b, l); 128 threads.
// ---------------------------------------------------------------------------
__global__ void density_kernel(const int* __restrict__ q,
                               const int* __restrict__ a,
                               const float* __restrict__ embed,
                               _Float16* __restrict__ dens) {
    const int blk = blockIdx.x;            // 0 .. 2*B*L-1, ordered (s, b, l)
    const int s   = blk / (Bsz * Lseq);
    const int rem = blk % (Bsz * Lseq);
    const int b   = rem / Lseq;
    const int l   = rem % Lseq;
    const int tid = threadIdx.x;           // 128 threads

    const int* tok = (s == 0) ? q : a;
    const int  t   = tok[b * Lseq + l];

    __shared__ float e[Dd];
    __shared__ float red[Dd];

    float v = embed[(size_t)t * Dd + tid];
    e[tid]   = v;
    red[tid] = v * v;
    __syncthreads();
    for (int off = Dd / 2; off > 0; off >>= 1) {
        if (tid < off) red[tid] += red[tid + off];
        __syncthreads();
    }
    const float inv = 1.0f / (red[0] + 1e-4f);
    const float ei  = e[tid] * inv;

    _Float16* out = dens + (size_t)blk * Dd * Dd + (size_t)tid * Dd;
#pragma unroll 8
    for (int j = 0; j < Dd; ++j) {
        out[j] = (_Float16)(ei * e[j]);
    }
}

// ---------------------------------------------------------------------------
// Kernel 2: one convLSTM layer, scanned over L timesteps.
// One persistent workgroup (1024 threads = 32 waves) per (seq, batch) pair.
//
// Conv input [x_t ; h] lives in a zero-padded, double-buffered LDS tile:
//   rows:  0 (zero) | 1..128 = x_t | 129..256 = h | 257 (zero)
//   cols:  0..7 zero pad (keeps interior 16B-aligned) | 8..135 data | pad
// => every im2col tap is a branch-free ds load with an immediate offset.
//
// Software-pipelined staging: x_{t+1} is DMA'd into the *next* buffer with
// global_load_async_to_lds_b128 while step t computes from the current one
// (the async writes touch rows 1..128 of next; compute writes h into rows
// 129..256 of next -> disjoint).  ASYNCcnt is drained once per step.
//
// Cell state c stays in registers; row/strip loops are fully unrolled so all
// creg indices are compile-time constants (no movrel).
//
// Gate convs via v_wmma_f32_16x16x32_f16:
//   A (16x32 f16) = gate weights, M = gate (4 of 16), K = tap p*3+r (12 of 32)
//   B (32x16 f16) = im2col patch (lanes 0-15 carry K=0..15; lanes 16-31 read
//                   the zero pad column so K=16..31 contribute nothing)
//   D (16x16 f32) = 4 gate pre-activations x 16 output columns.
// ---------------------------------------------------------------------------
constexpr int PadStride = 144;   // halfs per padded row (288 B, 16B-divisible)
constexpr int PadRows   = 258;   // rows -1..256
constexpr int BufHalfs  = PadRows * PadStride;   // 37152 halfs = 74304 B

__global__ __launch_bounds__(1024, 1) void convlstm_kernel(
        const _Float16* __restrict__ in,    // (2,B,L,D,D) f16
        _Float16* __restrict__ out,         // (2,B,L,D,D) f16
        const float* __restrict__ conv_w,   // (NL,4,1,1,4,3)
        const float* __restrict__ conv_b,   // (NL,4)
        int layer) {
    __shared__ _Float16 s_in[2][BufHalfs];  // ~145 KB total

    const int tid  = threadIdx.x;
    const int lane = tid & 31;              // wave32
    const int wave = tid >> 5;              // 0..31
    const int bs   = blockIdx.x;            // (s*B + b), 64 blocks

    const float* w    = conv_w + layer * 4 * 4 * 3;  // [g][p][r], 12 taps/gate
    const float* bias = conv_b + layer * 4;

    // --- Build A (weights) once: contiguous K order (K = 16*(lane/16) + h).
    v16h Amat;
    {
        const int m     = lane & 15;
        const int kbase = (lane < 16) ? 0 : 16;
#pragma unroll
        for (int h = 0; h < 16; ++h) {
            const int K = kbase + h;
            float wv = 0.0f;
            if (m < 4 && K < 12) wv = w[m * 12 + K];
            Amat[h] = (_Float16)wv;
        }
    }
    const float b0 = bias[0], b1 = bias[1], b2 = bias[2], b3 = bias[3];

    const size_t seqbase = (size_t)bs * Lseq * Dd * Dd;

    // Cell state in registers: wave owns rows [wave*4, wave*4+4),
    // lanes 0..15 own column js+lane of each 16-wide strip.
    float creg[4 * 8];
#pragma unroll
    for (int k = 0; k < 32; ++k) creg[k] = 0.0f;

    // Zero both padded buffers (padding must stay zero; h(t=0) = 0).
    {
        uint32_t* z = (uint32_t*)&s_in[0][0];
        const int nz = 2 * BufHalfs / 2;
        for (int i = tid; i < nz; i += blockDim.x) z[i] = 0u;
    }
    __syncthreads();

    // Async-stage x_t (128 rows x 256B = 2048 b128 chunks, 2 per thread)
    // into padded rows 1..128 of buffer `buf`.
    auto issue_x = [&](int t, int buf) {
        const _Float16* xp = in + seqbase + (size_t)t * Dd * Dd;
        _Float16* xdst = s_in[buf];
        for (int c = tid; c < 2048; c += blockDim.x) {
            const int row = c >> 4;
            const int seg = c & 15;
            const unsigned ldsb = (unsigned)(uintptr_t)(
                    xdst + (1 + row) * PadStride + 8 + seg * 8);
            const unsigned goff = (unsigned)(c * 16);
            asm volatile("global_load_async_to_lds_b128 %0, %1, %2"
                         :
                         : "v"(ldsb), "v"(goff), "s"(xp)
                         : "memory");
        }
    };

    // Prologue: stage x_0, drain, sync.
    issue_x(0, 0);
    asm volatile("s_wait_asynccnt 0" ::: "memory");
    __syncthreads();

    int cur = 0;
    for (int t = 0; t < Lseq; ++t) {
        // Kick off next timestep's DMA; it lands in disjoint LDS rows.
        if (t + 1 < Lseq) issue_x(t + 1, cur ^ 1);

        const _Float16* inb  = s_in[cur];
        _Float16*       nxt  = s_in[cur ^ 1];
        _Float16*       outp = out + seqbase + (size_t)t * Dd * Dd;

#pragma unroll
        for (int rr = 0; rr < 4; ++rr) {
            const int i = wave * 4 + rr;            // output row 0..127
#pragma unroll
            for (int st = 0; st < 8; ++st) {
                const int js = st * 16;
                // im2col base: lanes 0-15 -> tap window; lanes 16-31 -> the
                // always-zero left-pad column (their K=16..31 must be 0).
                const int colbase = (lane < 16) ? (8 + js + (lane & 15) - 1) : 0;
                const _Float16* bp = inb + (2 * i) * PadStride + colbase;
                v16h Bm;
#pragma unroll
                for (int h = 0; h < 12; ++h) {
                    Bm[h] = bp[(h / 3) * PadStride + (h % 3)];
                }
#pragma unroll
                for (int h = 12; h < 16; ++h) Bm[h] = (_Float16)0.0f;

                v8f acc = {};
                acc = __builtin_amdgcn_wmma_f32_16x16x32_f16(
                        /*neg_a=*/false, Amat, /*neg_b=*/false, Bm,
                        /*c_mod=*/(short)0, acc,
                        /*reuse_a=*/false, /*reuse_b=*/false);
                // D layout: VGPR g holds M=g on lanes 0-15 -> gates 0..3.
                if (lane < 16) {
                    const int j  = js + lane;
                    const int ci = rr * 8 + st;     // compile-time constant
                    const float fg = sigmoid_f(acc[0] + b0);
                    const float ig = sigmoid_f(acc[1] + b1);
                    const float og = sigmoid_f(acc[2] + b2);
                    const float cs = tanh_f(acc[3] + b3);
                    const float c  = fg * creg[ci] + ig * cs;
                    const float hh = og * tanh_f(c);
                    creg[ci] = c;
                    const _Float16 hf = (_Float16)hh;
                    nxt[(129 + i) * PadStride + 8 + j] = hf;  // h_{t+1}
                    outp[i * Dd + j] = hf;
                }
            }
        }
        // Drain this wave's DMA for x_{t+1}, then group-wide sync.
        asm volatile("s_wait_asynccnt 0" ::: "memory");
        __syncthreads();
        cur ^= 1;
    }
}

// ---------------------------------------------------------------------------
// Kernel 3: max-pool over L, linear (2 x 2*D*D), log-softmax. One block per b.
// ---------------------------------------------------------------------------
__global__ void head_kernel(const _Float16* __restrict__ feat, // (2,B,L,D*D)
                            const float* __restrict__ lin_w,   // (2, 2*D*D)
                            const float* __restrict__ lin_b,   // (2,)
                            float* __restrict__ outp) {        // (B, 2)
    const int b   = blockIdx.x;
    const int tid = threadIdx.x;            // 256 threads
    __shared__ float red0[256], red1[256];

    float acc0 = 0.0f, acc1 = 0.0f;
    for (int s = 0; s < 2; ++s) {
        const _Float16* base = feat + (size_t)(s * Bsz + b) * Lseq * Dd * Dd;
        for (int idx = tid; idx < Dd * Dd; idx += blockDim.x) {
            float m = -3.4e38f;
#pragma unroll 4
            for (int t = 0; t < Lseq; ++t) {
                m = fmaxf(m, (float)base[(size_t)t * Dd * Dd + idx]);
            }
            const int widx = s * Dd * Dd + idx;
            acc0 += m * lin_w[widx];                 // row 0
            acc1 += m * lin_w[2 * Dd * Dd + widx];   // row 1
        }
    }
    red0[tid] = acc0;
    red1[tid] = acc1;
    __syncthreads();
    for (int off = 128; off > 0; off >>= 1) {
        if (tid < off) { red0[tid] += red0[tid + off]; red1[tid] += red1[tid + off]; }
        __syncthreads();
    }
    if (tid == 0) {
        const float s0 = red0[0] + lin_b[0];
        const float s1 = red1[0] + lin_b[1];
        const float mx = fmaxf(s0, s1);
        const float lse = mx + logf(expf(s0 - mx) + expf(s1 - mx));
        outp[b * 2 + 0] = s0 - lse;
        outp[b * 2 + 1] = s1 - lse;
    }
}

// ---------------------------------------------------------------------------
// Launch: density -> convLSTM layer0 -> convLSTM layer1 -> head
// Workspace: [0, 80MB)   density / layer-1 output (ping-pong reuse)
//            [80MB,160MB) layer-0 output
// ---------------------------------------------------------------------------
extern "C" void kernel_launch(void* const* d_in, const int* in_sizes, int n_in,
                              void* d_out, int out_size, void* d_ws, size_t ws_size,
                              hipStream_t stream) {
    const int*   q      = (const int*)d_in[0];
    const int*   a      = (const int*)d_in[1];
    const float* embed  = (const float*)d_in[2];
    const float* conv_w = (const float*)d_in[3];
    const float* conv_b = (const float*)d_in[4];
    const float* lin_w  = (const float*)d_in[5];
    const float* lin_b  = (const float*)d_in[6];
    float*       out    = (float*)d_out;

    const size_t tensor_elems = (size_t)2 * Bsz * Lseq * Dd * Dd;  // 41.9M f16
    _Float16* dens = (_Float16*)d_ws;                 // also layer-1 output
    _Float16* bufA = dens + tensor_elems;             // layer-0 output

    density_kernel<<<2 * Bsz * Lseq, Dd, 0, stream>>>(q, a, embed, dens);
    convlstm_kernel<<<2 * Bsz, 1024, 0, stream>>>(dens, bufA, conv_w, conv_b, 0);
    convlstm_kernel<<<2 * Bsz, 1024, 0, stream>>>(bufA, dens, conv_w, conv_b, 1);
    head_kernel<<<Bsz, 256, 0, stream>>>(dens, lin_w, lin_b, out);
}
